// D2R_28011776704861
// MI455X (gfx1250) — compile-verified, matrix operand order checked by reading, not run on hardware
//
#include <hip/hip_runtime.h>

// DTW alignment: teacher [16,400,1024] expanded to student length [16,600,1024].
// K1: f32->f16 convert + row norms.
// K2: WMMA f16 GEMM -> euclidean distances, written DIAGONAL-MAJOR (d = i+j).
// K3: anti-diagonal DTW DP: coalesced diagonal reads/writes, LDS carries deps,
//     prefetch of next diagonal overlapped with the barrier.
// K4: backtrack (diagonal-major R) + scatter expansion.

typedef __attribute__((ext_vector_type(16))) _Float16 v16h;
typedef __attribute__((ext_vector_type(8)))  _Float16 v8h;
typedef __attribute__((ext_vector_type(4)))  _Float16 v4h;
typedef __attribute__((ext_vector_type(8)))  float    v8f;

#define T1 400
#define T2 600
#define T2P 608                 // padded to multiple of 16 for maskless WMMA K-loop
#define DIM 1024
#define NB 16
#define PATHLEN (T1 + T2 - 1)   // 999 anti-diagonals

// ---------------- K1: convert fp32 -> f16 row + squared row norm ----------------
__global__ void __launch_bounds__(256)
cvt_norm_kernel(const float* __restrict__ src, _Float16* __restrict__ dst,
                float* __restrict__ norms, int valid_rows, int rows_out) {
  int r = blockIdx.x;
  int b = blockIdx.y;
  int tid = threadIdx.x;
  _Float16* drow = dst + ((size_t)b * rows_out + r) * DIM;
  float s = 0.f;
  int e = tid * 4;                       // 256 threads * 4 = 1024
  if (r < valid_rows) {
    const float* srow = src + ((size_t)b * valid_rows + r) * DIM;
    float4 v = *(const float4*)(srow + e);
    v4h h = { (_Float16)v.x, (_Float16)v.y, (_Float16)v.z, (_Float16)v.w };
    *(v4h*)(drow + e) = h;
    s = v.x * v.x + v.y * v.y + v.z * v.z + v.w * v.w;
  } else {
    v4h h = {};                          // zero-pad rows 600..607
    *(v4h*)(drow + e) = h;
  }
  __shared__ float red[256];
  red[tid] = s;
  __syncthreads();
  for (int w = 128; w > 0; w >>= 1) {
    if (tid < w) red[tid] += red[tid + w];
    __syncthreads();
  }
  if (tid == 0) norms[(size_t)b * rows_out + r] = red[0];
}

// ---------------- K2: WMMA distance GEMM, diagonal-major output ----------------
// One wave per 16x32 output tile (two N-subtiles share one A fragment).
// 16-bit A-frag layout (16x32): lanes {0-15,16-31} are M=0-15 twice; halves 0-7
// hold K = 8*sel+0..7, halves 8-15 hold K = 16+8*sel+0..7 (sel = lane>>4).
// B-frag (32x16) mirrors it with N on lanes, so for A@B^T with both operands
// row-major along K the fragment loads use the identical pattern.
// Output element (m,n) -> Ddiag[(m+n)*T1 + m]  (scatter absorbed here, where
// 7.6k independent waves hide L2 latency, keeping the serial DP coalesced).
__global__ void __launch_bounds__(32)
dist_wmma_kernel(const _Float16* __restrict__ A,   // [NB, T1,  DIM]
                 const _Float16* __restrict__ Bm,  // [NB, T2P, DIM]
                 const float* __restrict__ aa, const float* __restrict__ bb,
                 float* __restrict__ Dd) {         // [NB, PATHLEN, T1]
  int mt = blockIdx.x, ntp = blockIdx.y, b = blockIdx.z;
  int lane = threadIdx.x;
  int sel  = lane >> 4;
  int lrow = lane & 15;
  const _Float16* pa  = A  + ((size_t)b * T1  + mt * 16 + lrow) * DIM + sel * 8;
  const _Float16* pb0 = Bm + ((size_t)b * T2P + (ntp * 2) * 16 + lrow) * DIM + sel * 8;
  const _Float16* pb1 = pb0 + (size_t)16 * DIM;
  v8f c0 = {}, c1 = {};
  #pragma unroll 2
  for (int k0 = 0; k0 < DIM; k0 += 32) {
    v8h alo = *(const v8h*)(pa  + k0);
    v8h ahi = *(const v8h*)(pa  + k0 + 16);
    v8h b0l = *(const v8h*)(pb0 + k0);
    v8h b0h = *(const v8h*)(pb0 + k0 + 16);
    v8h b1l = *(const v8h*)(pb1 + k0);
    v8h b1h = *(const v8h*)(pb1 + k0 + 16);
    v16h av  = __builtin_shufflevector(alo, ahi, 0,1,2,3,4,5,6,7,8,9,10,11,12,13,14,15);
    v16h bv0 = __builtin_shufflevector(b0l, b0h, 0,1,2,3,4,5,6,7,8,9,10,11,12,13,14,15);
    v16h bv1 = __builtin_shufflevector(b1l, b1h, 0,1,2,3,4,5,6,7,8,9,10,11,12,13,14,15);
    c0 = __builtin_amdgcn_wmma_f32_16x16x32_f16(false, av, false, bv0,
                                                (short)0, c0, false, false);
    c1 = __builtin_amdgcn_wmma_f32_16x16x32_f16(false, av, false, bv1,
                                                (short)0, c1, false, false);
  }
  // C/D layout: VGPR r -> M = mt*16 + sel*8 + r, N = ntile*16 + (lane&15)
  float* Db = Dd + (size_t)b * PATHLEN * T1;
  #pragma unroll
  for (int t = 0; t < 2; ++t) {
    int n = (ntp * 2 + t) * 16 + lrow;
    if (n < T2) {                         // padded columns have no diagonal slot
      float bbn = bb[(size_t)b * T2P + n];
      const v8f& c = t ? c1 : c0;
      #pragma unroll
      for (int r = 0; r < 8; ++r) {
        int m = mt * 16 + sel * 8 + r;
        float aam = aa[(size_t)b * T1 + m];
        float v = aam + bbn - 2.f * c[r];
        Db[(size_t)(m + n) * T1 + m] = sqrtf(fmaxf(v, 0.f));
      }
    }
  }
}

// ---------------- K3: DTW DP, anti-diagonal wavefront ----------------
// R[i,j] = D[i,j] + min(R[i-1,j], R[i,j-1], R[i-1,j-1]); in-place D->R in the
// diagonal-major buffer: fully coalesced per-diagonal global traffic. The two
// live previous diagonals stay in LDS; next diagonal is prefetched before the
// barrier so it sits in the WGP cache when the barrier releases.
__global__ void __launch_bounds__(512)
dtw_dp_kernel(float* __restrict__ Dd) {
  int b = blockIdx.x;
  int tid = threadIdx.x;
  float* Db = Dd + (size_t)b * PATHLEN * T1;
  __shared__ float sbuf[3][512];
  int cur = 0;
  for (int d = 0; d < PATHLEN; ++d) {
    float*       pc = sbuf[cur];             // diagonal d
    const float* p1 = sbuf[(cur + 2) % 3];   // diagonal d-1
    const float* p2 = sbuf[(cur + 1) % 3];   // diagonal d-2
    int i = tid;
    int j = d - i;
    if (i < T1 && j >= 0 && j < T2) {
      float dij = Db[(size_t)d * T1 + i];
      float v;
      if (i == 0 && j == 0)  v = dij;
      else if (i == 0)       v = dij + p1[0];        // left only
      else if (j == 0)       v = dij + p1[i - 1];    // up only
      else v = dij + fminf(fminf(p1[i - 1], p1[i]), p2[i - 1]);
      pc[i] = v;
      Db[(size_t)d * T1 + i] = v;
    }
    if (tid < T1 && d + 1 < PATHLEN)
      __builtin_prefetch(&Db[(size_t)(d + 1) * T1 + tid], 0, 3);
    __syncthreads();
    cur = (cur + 1) % 3;
  }
}

// ---------------- K4: backtrack + scatter expansion ----------------
// R(i,j) lives at Rd[(i+j)*T1 + i] (diagonal-major).
__global__ void __launch_bounds__(256)
backtrack_expand_kernel(const float* __restrict__ Rd,
                        const float* __restrict__ teacher,
                        float* __restrict__ out) {
  int b = blockIdx.x;
  int tid = threadIdx.x;
  const float* Rb = Rd + (size_t)b * PATHLEN * T1;
  __shared__ short pi[PATHLEN];
  __shared__ short pj[PATHLEN];
  if (tid == 0) {
    const float BIG = 1e30f;
    int i = T1 - 1, j = T2 - 1;
    bool done = false;
    for (int t = 0; t < PATHLEN; ++t) {
      if (!done) { pi[t] = (short)i; pj[t] = (short)j; }
      else       { pi[t] = 0;        pj[t] = -1;       }
      float cd = (i > 0 && j > 0) ? Rb[(size_t)(i + j - 2) * T1 + (i - 1)] : BIG;
      float cu = (i > 0)          ? Rb[(size_t)(i + j - 1) * T1 + (i - 1)] : BIG;
      float cl = (j > 0)          ? Rb[(size_t)(i + j - 1) * T1 + i]       : BIG;
      int ni = i, nj = j;
      if (cd <= cu && cd <= cl) { ni = i - 1; nj = j - 1; } // diag preferred on ties
      else if (cu <= cl)        { ni = i - 1; }             // then up
      else                      { nj = j - 1; }             // then left
      if (i == 0 && j == 0) { ni = 0; nj = 0; done = true; }
      i = ni; j = nj;
    }
  }
  __syncthreads();
  // j along the recorded path runs 599 -> 0 visiting every value; accumulate a
  // run of equal-j contributions in registers, write each output row once.
  const float* Tb = teacher + (size_t)b * T1 * DIM;
  float* Ob = out + (size_t)b * T2 * DIM;
  int e = tid * 4;
  float4 acc = make_float4(0.f, 0.f, 0.f, 0.f);
  int curj = T2 - 1;
  for (int t = 0; t < PATHLEN; ++t) {
    int j = pj[t];
    if (j < 0) break;                       // uniform across block
    int i = pi[t];
    if (j != curj) {
      *(float4*)(Ob + (size_t)curj * DIM + e) = acc;
      acc = make_float4(0.f, 0.f, 0.f, 0.f);
      curj = j;
    }
    float4 tv = *(const float4*)(Tb + (size_t)i * DIM + e);
    acc.x += tv.x; acc.y += tv.y; acc.z += tv.z; acc.w += tv.w;
  }
  *(float4*)(Ob + (size_t)curj * DIM + e) = acc;
}

extern "C" void kernel_launch(void* const* d_in, const int* in_sizes, int n_in,
                              void* d_out, int out_size, void* d_ws, size_t ws_size,
                              hipStream_t stream) {
  const float* teacher = (const float*)d_in[0];  // [16,400,1024] f32
  const float* student = (const float*)d_in[1];  // [16,600,1024] f32
  float* out = (float*)d_out;                    // [16,600,1024] f32

  char* ws = (char*)d_ws;
  _Float16* tf16 = (_Float16*)(ws);              // 16*400*1024*2  = 13,107,200 B
  _Float16* sf16 = (_Float16*)(ws + 13107200);   // 16*608*1024*2  = 19,922,944 B
  float*    aa   = (float*)(ws + 33030144);      // 16*400*4       =     25,600 B
  float*    bb   = (float*)(ws + 33055744);      // 16*608*4       =     38,912 B
  float*    Dd   = (float*)(ws + 33094656);      // 16*999*400*4   = 25,574,400 B
                                                 // total ~58.7 MB

  cvt_norm_kernel<<<dim3(T1, NB), 256, 0, stream>>>(teacher, tf16, aa, T1, T1);
  cvt_norm_kernel<<<dim3(T2P, NB), 256, 0, stream>>>(student, sf16, bb, T2, T2P);
  dist_wmma_kernel<<<dim3(T1 / 16, (T2P / 16) / 2, NB), 32, 0, stream>>>(tf16, sf16, aa, bb, Dd);
  dtw_dp_kernel<<<NB, 512, 0, stream>>>(Dd);
  backtrack_expand_kernel<<<NB, 256, 0, stream>>>(Dd, teacher, out);
}
